// HyperbolicRadialAttention_32693291057674
// MI455X (gfx1250) — compile-verified, matrix operand order checked by reading, not run on hardware
//
#include <hip/hip_runtime.h>
#include <hip/hip_bf16.h>
#include <math.h>

#define EMBED_DIM 1024
#define NUM_HEADS 16
#define HEAD_DIM  64
#define SEQ_LEN   2048
#define BATCH     2
#define NROWS     (BATCH * SEQ_LEN)   // 4096
#define EPSF      1e-8f

typedef __bf16 bf16_t;
typedef __attribute__((ext_vector_type(16))) __bf16 v16bf;
typedef __attribute__((ext_vector_type(8)))  __bf16 v8bf;
typedef __attribute__((ext_vector_type(8)))  float  v8f;

static __device__ inline v8f wmma_bf16(v16bf a, v16bf b, v8f c) {
  // D = A(16x32 bf16) * B(32x16 bf16) + C(16x16 f32)
  return __builtin_amdgcn_wmma_f32_16x16x32_bf16(
      /*neg_a=*/false, a, /*neg_b=*/false, b,
      /*c_mod=*/(short)0, c, /*reuse_a=*/false, /*reuse_b=*/false);
}

// Fragment loader for both A operands (lane&15 = matrix row, K contiguous in
// memory) and B operands whose K dim is contiguous per output column.
// ISA 7.12.2: lanes 0-15 hold K={0..7,16..23}, lanes 16-31 hold K={8..15,24..31}.
// stride multiple of 8 elements and k0 multiple of 8 keep 16B alignment.
static __device__ inline v16bf frag_ld(const bf16_t* p0, int stride, int k0, int lane) {
  int r   = lane & 15;
  int klo = k0 + ((lane >> 4) << 3);
  const bf16_t* p = p0 + r * stride + klo;
  v8bf lo = *(const v8bf*)(p);
  v8bf hi = *(const v8bf*)(p + 16);
  return __builtin_shufflevector(lo, hi, 0, 1, 2, 3, 4, 5, 6, 7,
                                 8, 9, 10, 11, 12, 13, 14, 15);
}

// ---------------------------------------------------------------------------
// fp32 -> bf16 pre-conversion (n multiple of 4)
// ---------------------------------------------------------------------------
__global__ __launch_bounds__(256)
void cvt_f32_to_bf16(const float* __restrict__ in, bf16_t* __restrict__ out, int n) {
  int i = (blockIdx.x * 256 + threadIdx.x) * 4;
  if (i < n) {
    float4 v = *(const float4*)(in + i);
    out[i + 0] = (bf16_t)v.x;
    out[i + 1] = (bf16_t)v.y;
    out[i + 2] = (bf16_t)v.z;
    out[i + 3] = (bf16_t)v.w;
  }
}

// ---------------------------------------------------------------------------
// GEMM: Y(MxN) = X(MxK)bf16 @ W(NxK)bf16^T + bias[N] (+ add(MxN))
// 128x128 block tile, 8 waves (4 row-groups x 2 col-groups), K step 32,
// double-buffered LDS (one barrier per k-step), next tile prefetched to regs,
// tile after that prefetched into cache.
// ---------------------------------------------------------------------------
__global__ __launch_bounds__(256)
void hyp_gemm_xwT(const bf16_t* __restrict__ X, const bf16_t* __restrict__ W,
                  const float* __restrict__ bias, const float* __restrict__ add,
                  float* __restrict__ Y, int M, int N, int K) {
  __shared__ bf16_t As[2][128][48];
  __shared__ bf16_t Bs[2][128][48];
  const int tid  = threadIdx.x;
  const int lane = tid & 31;
  const int wid  = tid >> 5;
  const int wr   = wid & 3;   // 4 groups of 32 rows
  const int wc   = wid >> 2;  // 2 groups of 64 cols
  const int m0 = blockIdx.y * 128;
  const int n0 = blockIdx.x * 128;

  const v8f vzero = {0.f, 0.f, 0.f, 0.f, 0.f, 0.f, 0.f, 0.f};
  v8f acc[2][4];
#pragma unroll
  for (int i = 0; i < 2; ++i)
#pragma unroll
    for (int j = 0; j < 4; ++j) acc[i][j] = vzero;

  // tile staging: 128 rows x 32 bf16; 2 threads/row, 16 bf16 (32B) each
  const int lrow = tid >> 1;
  const int lcol = (tid & 1) * 16;
  const bf16_t* xrow = X + lrow * K + m0 * K + lcol;
  const bf16_t* wrow = W + lrow * K + n0 * K + lcol;

  { // prologue: stage k0 = 0 into buffer 0
    uint4 a0 = ((const uint4*)xrow)[0], a1 = ((const uint4*)xrow)[1];
    uint4 b0 = ((const uint4*)wrow)[0], b1 = ((const uint4*)wrow)[1];
    uint4* da = (uint4*)&As[0][lrow][lcol];
    uint4* db = (uint4*)&Bs[0][lrow][lcol];
    da[0] = a0; da[1] = a1;
    db[0] = b0; db[1] = b1;
  }
  __syncthreads();

  int cur = 0;
  for (int k0 = 0; k0 < K; k0 += 32) {
    const bool has_next = (k0 + 32) < K;
    uint4 na0, na1, nb0, nb1;
    if (has_next) {  // issue next-tile global loads before the WMMA block
      const uint4* ap = (const uint4*)(xrow + k0 + 32);
      const uint4* bp = (const uint4*)(wrow + k0 + 32);
      na0 = ap[0]; na1 = ap[1];
      nb0 = bp[0]; nb1 = bp[1];
    }
    if (k0 + 64 < K) {  // cache prefetch two tiles ahead
      __builtin_prefetch(xrow + k0 + 64, 0, 1);
      __builtin_prefetch(wrow + k0 + 64, 0, 1);
    }
#pragma unroll
    for (int rt = 0; rt < 2; ++rt) {
      v16bf af = frag_ld(&As[cur][wr * 32 + rt * 16][0], 48, 0, lane);
#pragma unroll
      for (int ct = 0; ct < 4; ++ct) {
        v16bf bfrag = frag_ld(&Bs[cur][wc * 64 + ct * 16][0], 48, 0, lane);
        acc[rt][ct] = wmma_bf16(af, bfrag, acc[rt][ct]);
      }
    }
    if (has_next) {  // write other buffer; nobody reads it this step
      uint4* da = (uint4*)&As[cur ^ 1][lrow][lcol];
      uint4* db = (uint4*)&Bs[cur ^ 1][lrow][lcol];
      da[0] = na0; da[1] = na1;
      db[0] = nb0; db[1] = nb1;
    }
    __syncthreads();
    cur ^= 1;
  }

  // C tile layout: element i <-> row (i + 8*(lane>=16)), lane&15 <-> col.
  const int halfl = lane >> 4;
  const int nl    = lane & 15;
  if (add) {
#pragma unroll
    for (int rt = 0; rt < 2; ++rt)
#pragma unroll
      for (int ct = 0; ct < 4; ++ct) {
        int col = n0 + wc * 64 + ct * 16 + nl;
        float bcol = bias[col];
#pragma unroll
        for (int i = 0; i < 8; ++i) {
          int row = m0 + wr * 32 + rt * 16 + i + 8 * halfl;
          int idx = row * N + col;
          Y[idx] = acc[rt][ct][i] + bcol + add[idx];
        }
      }
  } else {
#pragma unroll
    for (int rt = 0; rt < 2; ++rt)
#pragma unroll
      for (int ct = 0; ct < 4; ++ct) {
        int col = n0 + wc * 64 + ct * 16 + nl;
        float bcol = bias[col];
#pragma unroll
        for (int i = 0; i < 8; ++i) {
          int row = m0 + wr * 32 + rt * 16 + i + 8 * halfl;
          Y[row * N + col] = acc[rt][ct][i] + bcol;
        }
      }
  }
}

// ---------------------------------------------------------------------------
// Hyperbolic projection + per-row squared norms. One wave per (token, head).
// ---------------------------------------------------------------------------
__global__ __launch_bounds__(256)
void hyp_project(const float* __restrict__ Qf, const float* __restrict__ Kf,
                 const float* __restrict__ Vf,
                 const float* __restrict__ curv, const float* __restrict__ rw,
                 bf16_t* __restrict__ qb, bf16_t* __restrict__ kb, bf16_t* __restrict__ vb,
                 float* __restrict__ qn2, float* __restrict__ kn2) {
  const int lane = threadIdx.x & 31;
  const int wid  = threadIdx.x >> 5;
  const int p = blockIdx.x * 8 + wid;   // (token,head) pair
  const int n = p >> 4;                 // token row in (B*L)
  const int h = p & 15;
  const int b = n / SEQ_LEN;
  const int l = n % SEQ_LEN;

  const float sc     = sqrtf(fmaxf(fabsf(curv[0]), EPSF));
  const float radial = 1.0f / (1.0f + __expf(-rw[h]));

  const int src    = n * EMBED_DIM + h * HEAD_DIM + lane * 2;
  const int dstrow = (b * NUM_HEADS + h) * SEQ_LEN + l;
  const int dst    = dstrow * HEAD_DIM + lane * 2;

  float2 xq = *(const float2*)(Qf + src);
  float2 xk = *(const float2*)(Kf + src);
  float2 xv = *(const float2*)(Vf + src);
  float nq = xq.x * xq.x + xq.y * xq.y;
  float nk = xk.x * xk.x + xk.y * xk.y;
  float nv = xv.x * xv.x + xv.y * xv.y;
#pragma unroll
  for (int m = 16; m >= 1; m >>= 1) {
    nq += __shfl_xor(nq, m, 32);
    nk += __shfl_xor(nk, m, 32);
    nv += __shfl_xor(nv, m, 32);
  }
  float aq = sc * sqrtf(nq), ak = sc * sqrtf(nk), av = sc * sqrtf(nv);
  float sq = (tanhf(aq) / (aq + EPSF)) * radial;  // radial applied to q
  float sk =  tanhf(ak) / (ak + EPSF);
  float sv =  tanhf(av) / (av + EPSF);

  qb[dst]     = (bf16_t)(xq.x * sq);
  qb[dst + 1] = (bf16_t)(xq.y * sq);
  kb[dst]     = (bf16_t)(xk.x * sk);
  kb[dst + 1] = (bf16_t)(xk.y * sk);
  vb[dst]     = (bf16_t)(xv.x * sv);
  vb[dst + 1] = (bf16_t)(xv.y * sv);
  if (lane == 0) {
    qn2[dstrow] = sq * sq * nq;
    kn2[dstrow] = sk * sk * nk;
  }
}

// ---------------------------------------------------------------------------
// Flash-style hyperbolic attention, transposed formulation.
//   S^T = K @ Q^T   (A = K tile from LDS, B = Q rows; C-layout: lane = query,
//                    element = key)  ->  softmax stats are per-lane scalars.
//   O^T = V^T @ P^T (A = transposed V from LDS; B operand built DIRECTLY from
//                    two S^T C-tiles in registers -- no LDS round trip:
//                    B-frag element pattern {0..7,16..23}/{8..15,24..31}
//                    matches stacked C-tile rows exactly.)
// Block = 128 query rows of one (b,h); 8 waves x 16 queries.
// ---------------------------------------------------------------------------
__global__ __launch_bounds__(256)
void hyp_attention(const bf16_t* __restrict__ qb, const bf16_t* __restrict__ kb,
                   const bf16_t* __restrict__ vb,
                   const float* __restrict__ qn2, const float* __restrict__ kn2,
                   const float* __restrict__ curv,
                   bf16_t* __restrict__ out) {
  __shared__ bf16_t Ks[64][80];   // K chunk, row-major [s][d]
  __shared__ bf16_t Vt[64][80];   // V chunk, transposed [d][s]
  __shared__ float  kn2s[64];     // key norms for the chunk

  const int tid  = threadIdx.x;
  const int lane = tid & 31;
  const int wid  = tid >> 5;
  const int bh = blockIdx.x >> 4;     // b*H + h   (L/128 == 16 q-tiles)
  const int qt = blockIdx.x & 15;
  const int b  = bh >> 4;
  const int h  = bh & 15;
  const int qrow0 = qt * 128 + wid * 16;

  const bf16_t* Qh  = qb  + bh * SEQ_LEN * HEAD_DIM;
  const bf16_t* Kh  = kb  + bh * SEQ_LEN * HEAD_DIM;
  const bf16_t* Vh  = vb  + bh * SEQ_LEN * HEAD_DIM;
  const float* qn2h = qn2 + bh * SEQ_LEN;
  const float* kn2h = kn2 + bh * SEQ_LEN;

  const float sc       = sqrtf(fmaxf(fabsf(curv[0]), EPSF));
  const float negscale = -1.0f / ((sc + EPSF) * 8.0f);  // 8 = sqrt(dh)

  const int halfl = lane >> 4;
  const int nl    = lane & 15;

  // Q^T B-fragments: lane = query column, elements = d (contiguous per row)
  v16bf bq0 = frag_ld(Qh + qrow0 * HEAD_DIM, HEAD_DIM, 0,  lane);
  v16bf bq1 = frag_ld(Qh + qrow0 * HEAD_DIM, HEAD_DIM, 32, lane);

  const float myqn2 = qn2h[qrow0 + nl];      // per-lane query norm
  const float qden  = 1.0f - myqn2;
  float mrow = -1e30f, lrow = 0.0f;          // per-lane softmax stats

  const v8f vzero = {0.f, 0.f, 0.f, 0.f, 0.f, 0.f, 0.f, 0.f};
  v8f accO[4];                               // O^T: element = d, lane = query
#pragma unroll
  for (int dt = 0; dt < 4; ++dt) accO[dt] = vzero;

  // staging roles: 4 threads per 64-wide row, 16 bf16 (2x uint4) each
  const int srow = tid >> 2;
  const int scol = (tid & 3) * 16;

  for (int s0 = 0; s0 < SEQ_LEN; s0 += 64) {
    __syncthreads();  // previous chunk's fragment reads are complete
    {
      const uint4* kp = (const uint4*)(Kh + (s0 + srow) * HEAD_DIM + scol);
      uint4 k0v = kp[0], k1v = kp[1];
      uint4* kd = (uint4*)&Ks[srow][scol];
      kd[0] = k0v; kd[1] = k1v;

      const bf16_t* vp = Vh + (s0 + srow) * HEAD_DIM + scol;
      v8bf v0 = *(const v8bf*)(vp);
      v8bf v1 = *(const v8bf*)(vp + 8);
#pragma unroll
      for (int j = 0; j < 8; ++j) {   // transpose: Vt[d][s]
        Vt[scol + j][srow]     = v0[j];
        Vt[scol + 8 + j][srow] = v1[j];
      }
      if (tid < 64) kn2s[tid] = kn2h[s0 + tid];
    }
    __syncthreads();

    // ---- S^T = K Q^T : element = key row, lane = query col ----
    v8f st[4];
#pragma unroll
    for (int nt = 0; nt < 4; ++nt) {
      v16bf ak0 = frag_ld(&Ks[nt * 16][0], 80, 0,  lane);
      v16bf ak1 = frag_ld(&Ks[nt * 16][0], 80, 32, lane);
      v8f z = vzero;
      z = wmma_bf16(ak0, bq0, z);
      z = wmma_bf16(ak1, bq1, z);
      st[nt] = z;
    }

    // ---- hyperbolic distance -> scores (kn2 per element via LDS v8f) ----
#pragma unroll
    for (int nt = 0; nt < 4; ++nt) {
      v8f kvec = *(const v8f*)&kn2s[nt * 16 + 8 * halfl];
#pragma unroll
      for (int i = 0; i < 8; ++i) {
        float kn    = kvec[i];
        float d2    = myqn2 + kn - 2.0f * st[nt][i];
        float denom = fmaxf(qden * (1.0f - kn), EPSF);
        float x     = fmaxf(1.0f + 2.0f * d2 / denom, 1.0f + EPSF);
        float dist  = __logf(x + sqrtf(fmaxf(x * x - 1.0f, 0.0f)));  // arccosh
        st[nt][i]   = dist * negscale;
      }
    }

    // ---- online softmax: in-lane over 32 keys + one half-merge shuffle ----
    float t = st[0][0];
#pragma unroll
    for (int nt = 0; nt < 4; ++nt)
#pragma unroll
      for (int i = 0; i < 8; ++i) t = fmaxf(t, st[nt][i]);
    t = fmaxf(t, __shfl_xor(t, 16, 32));
    const float mnew  = fmaxf(mrow, t);
    const float alpha = __expf(mrow - mnew);

    float s = 0.0f;
#pragma unroll
    for (int nt = 0; nt < 4; ++nt)
#pragma unroll
      for (int i = 0; i < 8; ++i) {
        float e = __expf(st[nt][i] - mnew);
        st[nt][i] = e;
        s += e;
      }
    s += __shfl_xor(s, 16, 32);
    lrow = lrow * alpha + s;
    mrow = mnew;
#pragma unroll
    for (int dt = 0; dt < 4; ++dt) accO[dt] *= alpha;

    // ---- O^T += V^T P^T : B operand packed straight from C tiles ----
#pragma unroll
    for (int ks = 0; ks < 2; ++ks) {
      v16bf bp;
#pragma unroll
      for (int i = 0; i < 8; ++i) {
        bp[i]     = (bf16_t)st[2 * ks][i];
        bp[8 + i] = (bf16_t)st[2 * ks + 1][i];
      }
#pragma unroll
      for (int dt = 0; dt < 4; ++dt) {
        v16bf av = frag_ld(&Vt[dt * 16][0], 80, ks * 32, lane);
        accO[dt] = wmma_bf16(av, bp, accO[dt]);
      }
    }
  }

  // ---- normalize, write bf16 in (B, L, D) layout for the final GEMM ----
  const float invl = 1.0f / lrow;
  const int row = b * SEQ_LEN + qrow0 + nl;
#pragma unroll
  for (int dt = 0; dt < 4; ++dt)
#pragma unroll
    for (int i = 0; i < 8; ++i) {
      int col = h * HEAD_DIM + dt * 16 + i + 8 * halfl;
      out[row * EMBED_DIM + col] = (bf16_t)(accO[dt][i] * invl);
    }
}

// ---------------------------------------------------------------------------
extern "C" void kernel_launch(void* const* d_in, const int* in_sizes, int n_in,
                              void* d_out, int out_size, void* d_ws, size_t ws_size,
                              hipStream_t stream) {
  const float* query = (const float*)d_in[0];
  const float* Wq    = (const float*)d_in[1];
  const float* bq    = (const float*)d_in[2];
  const float* Wk    = (const float*)d_in[3];
  const float* bk    = (const float*)d_in[4];
  const float* Wv    = (const float*)d_in[5];
  const float* bv    = (const float*)d_in[6];
  const float* Wo    = (const float*)d_in[7];
  const float* bo    = (const float*)d_in[8];
  const float* Wb    = (const float*)d_in[9];
  const float* bb    = (const float*)d_in[10];
  const float* curv  = (const float*)d_in[11];
  const float* rw    = (const float*)d_in[12];
  float* out = (float*)d_out;

  char* ws = (char*)d_ws;
  size_t off = 0;
  auto carve = [&](size_t bytes) {
    void* p = ws + off;
    off += (bytes + 255) & ~(size_t)255;
    return p;
  };
  const size_t matBytes = (size_t)NROWS * EMBED_DIM * sizeof(float);      // 16 MB
  const size_t bfBytes  = (size_t)NROWS * EMBED_DIM * sizeof(bf16_t);     // 8 MB
  const size_t wBytes   = (size_t)EMBED_DIM * EMBED_DIM * sizeof(bf16_t); // 2 MB
  const size_t nrmBytes = (size_t)BATCH * NUM_HEADS * SEQ_LEN * sizeof(float);

  bf16_t* Xbf  = (bf16_t*)carve(bfBytes);
  bf16_t* Wqb  = (bf16_t*)carve(wBytes);
  bf16_t* Wkb  = (bf16_t*)carve(wBytes);
  bf16_t* Wvb  = (bf16_t*)carve(wBytes);
  bf16_t* Wbb  = (bf16_t*)carve(wBytes);
  bf16_t* Wob  = (bf16_t*)carve(wBytes);
  float*  Qf   = (float*)carve(matBytes);
  float*  Kf   = (float*)carve(matBytes);
  float*  Vf   = (float*)carve(matBytes);
  float*  Bres = (float*)carve(matBytes);
  bf16_t* qbuf = (bf16_t*)carve(bfBytes);
  bf16_t* kbuf = (bf16_t*)carve(bfBytes);
  bf16_t* vbuf = (bf16_t*)carve(bfBytes);
  bf16_t* aOut = (bf16_t*)carve(bfBytes);
  float*  qn2  = (float*)carve(nrmBytes);
  float*  kn2  = (float*)carve(nrmBytes);

  const int nX = NROWS * EMBED_DIM;       // 4 M
  const int nW = EMBED_DIM * EMBED_DIM;   // 1 M
  cvt_f32_to_bf16<<<nX / 1024, 256, 0, stream>>>(query, Xbf, nX);
  cvt_f32_to_bf16<<<nW / 1024, 256, 0, stream>>>(Wq, Wqb, nW);
  cvt_f32_to_bf16<<<nW / 1024, 256, 0, stream>>>(Wk, Wkb, nW);
  cvt_f32_to_bf16<<<nW / 1024, 256, 0, stream>>>(Wv, Wvb, nW);
  cvt_f32_to_bf16<<<nW / 1024, 256, 0, stream>>>(Wb, Wbb, nW);
  cvt_f32_to_bf16<<<nW / 1024, 256, 0, stream>>>(Wo, Wob, nW);

  dim3 gGemm(EMBED_DIM / 128, NROWS / 128);  // (8, 32)
  hyp_gemm_xwT<<<gGemm, 256, 0, stream>>>(Xbf, Wqb, bq, nullptr, Qf,   NROWS, EMBED_DIM, EMBED_DIM);
  hyp_gemm_xwT<<<gGemm, 256, 0, stream>>>(Xbf, Wkb, bk, nullptr, Kf,   NROWS, EMBED_DIM, EMBED_DIM);
  hyp_gemm_xwT<<<gGemm, 256, 0, stream>>>(Xbf, Wvb, bv, nullptr, Vf,   NROWS, EMBED_DIM, EMBED_DIM);
  hyp_gemm_xwT<<<gGemm, 256, 0, stream>>>(Xbf, Wbb, bb, nullptr, Bres, NROWS, EMBED_DIM, EMBED_DIM);

  hyp_project<<<(NROWS * NUM_HEADS) / 8, 256, 0, stream>>>(
      Qf, Kf, Vf, curv, rw, qbuf, kbuf, vbuf, qn2, kn2);

  hyp_attention<<<BATCH * NUM_HEADS * (SEQ_LEN / 128), 256, 0, stream>>>(
      qbuf, kbuf, vbuf, qn2, kn2, curv, aOut);

  hyp_gemm_xwT<<<gGemm, 256, 0, stream>>>(aOut, Wob, bo, Bres, out, NROWS, EMBED_DIM, EMBED_DIM);
}
